// BitNet_876173328457
// MI455X (gfx1250) — 16-bit and 8-bit matrix operand order repaired and checked by reading, compiled, not run on hardware
//
#include <hip/hip_runtime.h>
#include <hip/hip_bf16.h>
#include <cstdint>


typedef __attribute__((ext_vector_type(4)))  int      v4i;
typedef __attribute__((ext_vector_type(8)))  int      v8i;
typedef __attribute__((ext_vector_type(8)))  float    v8f;
typedef __attribute__((ext_vector_type(16))) _Float16 v16h;

constexpr int S_  = 2048;
constexpr int H_  = 2560;
constexpr int NH_ = 20;
constexpr int NKV_= 5;
constexpr int HD_ = 128;
constexpr int FF_ = 6912;

#ifndef __has_builtin
#define __has_builtin(x) 0
#endif

#if defined(__HIP_DEVICE_COMPILE__) && __has_builtin(__builtin_amdgcn_global_load_async_to_lds_b128)
#define ASYNC_LDS 1
#else
#define ASYNC_LDS 0
#endif

__device__ inline void copy16_g2l(const int8_t* g, int8_t* l)
{
#if ASYNC_LDS
  __builtin_amdgcn_global_load_async_to_lds_b128(
      (__attribute__((address_space(1))) v4i*)g,
      (__attribute__((address_space(3))) v4i*)l, 0, 0);
#else
  *(uint4*)l = *(const uint4*)g;
#endif
}
__device__ inline void async_wait_le1()
{
#if ASYNC_LDS
#if __has_builtin(__builtin_amdgcn_s_wait_asynccnt)
  __builtin_amdgcn_s_wait_asynccnt(1);
#else
  asm volatile("s_wait_asynccnt 1" ::: "memory");
#endif
#endif
}
__device__ inline void async_wait_le0()
{
#if ASYNC_LDS
#if __has_builtin(__builtin_amdgcn_s_wait_asynccnt)
  __builtin_amdgcn_s_wait_asynccnt(0);
#else
  asm volatile("s_wait_asynccnt 0" ::: "memory");
#endif
#endif
}

__device__ inline v8f v8f_zero() {
  v8f z;
  #pragma unroll
  for (int i = 0; i < 8; i++) z[i] = 0.f;
  return z;
}
__device__ inline v8i v8i_zero() {
  v8i z;
  #pragma unroll
  for (int i = 0; i < 8; i++) z[i] = 0;
  return z;
}

union U8pack { v8i v; uint2 d2[4]; uint4 d4[2]; };

__device__ inline void mma4_step(const int8_t* lbase, const U8pack& a, v8i acc[4])
{
  U8pack b[4];
  #pragma unroll
  for (int t = 0; t < 4; t++) {
    b[t].d2[0] = *(const uint2*)(lbase + t*1024);
    b[t].d2[1] = *(const uint2*)(lbase + t*1024 + 16);
    b[t].d2[2] = *(const uint2*)(lbase + t*1024 + 32);
    b[t].d2[3] = *(const uint2*)(lbase + t*1024 + 48);
  }
  #pragma unroll
  for (int t = 0; t < 4; t++)
    acc[t] = __builtin_amdgcn_wmma_i32_16x16x64_iu8(
        true, a.v, true, b[t].v, acc[t], false, false);
}

__global__ __launch_bounds__(256) void k_abssum(const float* __restrict__ w, long n,
                                                float* __restrict__ acc)
{
  __shared__ float red[256];
  float s = 0.f;
  for (long i = (long)blockIdx.x*blockDim.x + threadIdx.x; i < n;
       i += (long)gridDim.x*blockDim.x)
    s += fabsf(w[i]);
  red[threadIdx.x] = s; __syncthreads();
  for (int o = 128; o > 0; o >>= 1) {
    if ((int)threadIdx.x < o) red[threadIdx.x] += red[threadIdx.x+o];
    __syncthreads();
  }
  if (threadIdx.x == 0) atomicAdd(acc, red[0]);
}

__global__ __launch_bounds__(256) void k_wquant(const float* __restrict__ w, long n,
                                                const float* __restrict__ acc, float invn,
                                                int8_t* __restrict__ wq)
{
  float ws  = fmaxf(acc[0]*invn, 1e-5f);
  float inv = 1.f/ws;
  for (long i = (long)blockIdx.x*blockDim.x + threadIdx.x; i < n;
       i += (long)gridDim.x*blockDim.x) {
    float r = rintf(w[i]*inv);
    r = fminf(1.f, fmaxf(-1.f, r));
    wq[i] = (int8_t)r;
  }
}

__global__ __launch_bounds__(256) void k_rmsq(const float* __restrict__ x,
                                              const float* __restrict__ g, int C,
                                              int8_t* __restrict__ xq,
                                              float* __restrict__ srow)
{
  __shared__ float rs[256], rm[256];
  __shared__ float s_scale, s_qs;
  int row = blockIdx.x;
  const float* xr = x + (size_t)row*C;
  float ss = 0.f, mx = 0.f;
  for (int c = threadIdx.x; c < C; c += 256) {
    float v = xr[c];
    ss += v*v;
    mx = fmaxf(mx, fabsf(v*g[c]));
  }
  rs[threadIdx.x] = ss; rm[threadIdx.x] = mx; __syncthreads();
  for (int o = 128; o > 0; o >>= 1) {
    if ((int)threadIdx.x < o) {
      rs[threadIdx.x] += rs[threadIdx.x+o];
      rm[threadIdx.x]  = fmaxf(rm[threadIdx.x], rm[threadIdx.x+o]);
    }
    __syncthreads();
  }
  if (threadIdx.x == 0) {
    float scale = rsqrtf(rs[0]/(float)C + 1e-5f);
    float s     = fmaxf(scale*rm[0], 1e-5f);
    srow[row] = s; s_scale = scale; s_qs = 127.f/s;
  }
  __syncthreads();
  float scale = s_scale, qs = s_qs;
  int8_t* xo = xq + (size_t)row*C;
  for (int c = threadIdx.x; c < C; c += 256) {
    float q = rintf(xr[c]*g[c]*scale*qs);
    q = fminf(127.f, fmaxf(-128.f, q));
    xo[c] = (int8_t)q;
  }
}

__global__ __launch_bounds__(256) void k_gemm_q8(
    const int8_t* __restrict__ Aq, const float* __restrict__ srow,
    const int8_t* __restrict__ Wq, const float* __restrict__ wsacc, float winvn,
    float* __restrict__ out, int M, int N, int K)
{
  __shared__ __align__(16) int8_t ldsB[2][64*64];
  int tid  = threadIdx.x;
  int wv   = tid >> 5, lane = tid & 31;
  int l16  = lane & 15, hi = lane >> 4;
  int nTn  = N >> 6;
  int bm   = blockIdx.x / nTn, bn = blockIdx.x % nTn;
  int m0   = bm*128 + wv*16, n0 = bn*64;

  int crow = tid >> 2, cseg = (tid & 3) << 4;
  const int8_t* gB  = Wq + (size_t)(n0 + crow)*K + cseg;
  int8_t*       lB  = &ldsB[0][0] + crow*64 + cseg;
  int8_t*       lB1 = &ldsB[1][0] + crow*64 + cseg;

  v8i acc[4];
  #pragma unroll
  for (int t = 0; t < 4; t++) acc[t] = v8i_zero();

  const int8_t* arow = Aq + (size_t)(m0 + l16)*K + hi*8;
  const int lboff = l16*64 + hi*8;

  copy16_g2l(gB, lB);
  U8pack a;
  a.d2[0] = *(const uint2*)(arow +  0);
  a.d2[1] = *(const uint2*)(arow + 16);
  a.d2[2] = *(const uint2*)(arow + 32);
  a.d2[3] = *(const uint2*)(arow + 48);

  int cur = 0;
  int k0  = 0;
  for (; k0 < K - 64; k0 += 64) {
    copy16_g2l(gB + k0 + 64, (cur ? lB : lB1));
    async_wait_le1();
    __syncthreads();

    U8pack anext;
    const int8_t* ap = arow + k0 + 64;
    anext.d2[0] = *(const uint2*)(ap +  0);
    anext.d2[1] = *(const uint2*)(ap + 16);
    anext.d2[2] = *(const uint2*)(ap + 32);
    anext.d2[3] = *(const uint2*)(ap + 48);
    __builtin_prefetch(ap + 64, 0, 1);

    mma4_step(&ldsB[cur][0] + lboff, a, acc);

    __syncthreads();
    a = anext; cur ^= 1;
  }
  async_wait_le0();
  __syncthreads();
  mma4_step(&ldsB[cur][0] + lboff, a, acc);

  float ws = fmaxf(wsacc[0]*winvn, 1e-5f) * (1.f/127.f);
  float rowsc[8];
  #pragma unroll
  for (int i = 0; i < 8; i++) rowsc[i] = srow[m0 + hi*8 + i] * ws;
  #pragma unroll
  for (int t = 0; t < 4; t++) {
    #pragma unroll
    for (int i = 0; i < 8; i++) {
      int m = m0 + hi*8 + i;
      int n = n0 + t*16 + l16;
      out[(size_t)m*N + n] = (float)acc[t][i] * rowsc[i];
    }
  }
}

__global__ __launch_bounds__(256) void k_rope(const float* __restrict__ xin,
                                              const float* __restrict__ cosb,
                                              const float* __restrict__ sinb,
                                              _Float16* __restrict__ xout, int nh)
{
  long idx = (long)blockIdx.x*blockDim.x + threadIdx.x;
  long total = (long)nh*S_*HD_;
  if (idx >= total) return;
  int d = (int)(idx % HD_);
  long t = idx / HD_;
  int s = (int)(t % S_);
  int h = (int)(t / S_);
  size_t base = (size_t)s*nh*HD_ + (size_t)h*HD_;
  float x  = xin[base + d];
  float xr = (d < HD_/2) ? -xin[base + d + HD_/2] : xin[base + d - HD_/2];
  float c  = cosb[(size_t)s*HD_ + d];
  float sn = sinb[(size_t)s*HD_ + d];
  xout[idx] = (_Float16)(x*c + xr*sn);
}

__global__ __launch_bounds__(256) void k_vt(const float* __restrict__ vin,
                                            _Float16* __restrict__ vout)
{
  long idx = (long)blockIdx.x*blockDim.x + threadIdx.x;
  long total = (long)NKV_*HD_*S_;
  if (idx >= total) return;
  int s = (int)(idx % S_);
  long t = idx / S_;
  int d = (int)(t % HD_);
  int h = (int)(t / HD_);
  vout[idx] = (_Float16)vin[(size_t)s*NKV_*HD_ + (size_t)h*HD_ + d];
}

__global__ __launch_bounds__(256) void k_flash(
    const _Float16* __restrict__ Q,
    const _Float16* __restrict__ Kt,
    const _Float16* __restrict__ Vt,
    float* __restrict__ O)
{
  __shared__ __align__(16) _Float16 sP[8][16][32];
  int wavein = threadIdx.x >> 5;
  int wave   = blockIdx.x*8 + wavein;
  int lane   = threadIdx.x & 31;
  int l16 = lane & 15, hi = lane >> 4;
  int nqt = S_ >> 4;
  if (wave >= nqt*NH_) return;
  int h  = wave / nqt;
  int q0 = (wave % nqt) << 4;
  int kvh = h / (NH_/NKV_);
  const float scl = 0.08838834764831845f;

  union HU { v16h v; uint4 u[2]; };

  HU qa[4];
  const _Float16* qb = Q + ((size_t)h*S_ + q0 + l16)*HD_;
  #pragma unroll
  for (int kc = 0; kc < 4; kc++) {
    qa[kc].u[0] = *(const uint4*)(qb + kc*32 +      hi*8);
    qa[kc].u[1] = *(const uint4*)(qb + kc*32 + 16 + hi*8);
  }

  v8f oacc[8];
  #pragma unroll
  for (int t = 0; t < 8; t++) oacc[t] = v8f_zero();
  float mrow[8], lsum[8];
  #pragma unroll
  for (int i = 0; i < 8; i++) { mrow[i] = -1e30f; lsum[i] = 0.f; }

  for (int jb = 0; jb < q0 + 16; jb += 32) {
    v8f s0 = v8f_zero(), s1 = v8f_zero();
    const _Float16* kbase = Kt + (size_t)kvh*S_*HD_;
    #pragma unroll
    for (int kc = 0; kc < 4; kc++) {
      HU kb0, kb1;
      const _Float16* p0 = kbase + (size_t)(jb      + l16)*HD_ + kc*32 + hi*8;
      const _Float16* p1 = kbase + (size_t)(jb + 16 + l16)*HD_ + kc*32 + hi*8;
      kb0.u[0] = ((const uint4*)p0)[0]; kb0.u[1] = ((const uint4*)p0)[2];
      kb1.u[0] = ((const uint4*)p1)[0]; kb1.u[1] = ((const uint4*)p1)[2];
      s0 = __builtin_amdgcn_wmma_f32_16x16x32_f16(false, qa[kc].v, false, kb0.v,
                                                  (short)0, s0, false, false);
      s1 = __builtin_amdgcn_wmma_f32_16x16x32_f16(false, qa[kc].v, false, kb1.v,
                                                  (short)0, s1, false, false);
    }
    float p0[8], p1[8], alpha[8];
    #pragma unroll
    for (int i = 0; i < 8; i++) {
      int qrow = q0 + hi*8 + i;
      int k0i = jb + l16, k1i = jb + 16 + l16;
      float v0 = (k0i <= qrow) ? s0[i]*scl : -1e30f;
      float v1 = (k1i <= qrow) ? s1[i]*scl : -1e30f;
      float rmax = fmaxf(v0, v1);
      #pragma unroll
      for (int msk = 1; msk < 16; msk <<= 1)
        rmax = fmaxf(rmax, __shfl_xor(rmax, msk, 32));
      float mnew = fmaxf(mrow[i], rmax);
      alpha[i] = __expf(mrow[i] - mnew);
      mrow[i]  = mnew;
      float e0 = __expf(v0 - mnew), e1 = __expf(v1 - mnew);
      float rs = e0 + e1;
      #pragma unroll
      for (int msk = 1; msk < 16; msk <<= 1)
        rs += __shfl_xor(rs, msk, 32);
      lsum[i] = lsum[i]*alpha[i] + rs;
      p0[i] = e0; p1[i] = e1;
    }
    #pragma unroll
    for (int i = 0; i < 8; i++) {
      sP[wavein][hi*8 + i][l16]      = (_Float16)p0[i];
      sP[wavein][hi*8 + i][16 + l16] = (_Float16)p1[i];
    }
    HU pa;
    pa.u[0] = *(const uint4*)&sP[wavein][l16][hi*8];
    pa.u[1] = *(const uint4*)&sP[wavein][l16][16 + hi*8];
    #pragma unroll
    for (int t = 0; t < 8; t++) {
      const _Float16* vp = Vt + ((size_t)kvh*HD_ + t*16 + l16)*S_ + jb + hi*8;
      HU vb; vb.u[0] = ((const uint4*)vp)[0]; vb.u[1] = ((const uint4*)vp)[2];
      v8f c = oacc[t];
      #pragma unroll
      for (int i = 0; i < 8; i++) c[i] *= alpha[i];
      oacc[t] = __builtin_amdgcn_wmma_f32_16x16x32_f16(false, pa.v, false, vb.v,
                                                       (short)0, c, false, false);
    }
  }
  #pragma unroll
  for (int t = 0; t < 8; t++) {
    #pragma unroll
    for (int i = 0; i < 8; i++) {
      int r = q0 + hi*8 + i;
      int c = h*HD_ + t*16 + l16;
      O[(size_t)r*(NH_*HD_) + c] = oacc[t][i] / lsum[i];
    }
  }
}

__global__ __launch_bounds__(256) void k_add(const float* __restrict__ a,
                                             const float* __restrict__ b,
                                             float* __restrict__ o, long n)
{
  long i = (long)blockIdx.x*blockDim.x + threadIdx.x;
  if (i < n) o[i] = a[i] + b[i];
}

__global__ __launch_bounds__(256) void k_sqrelu_mul(float* __restrict__ g,
                                                    const float* __restrict__ u, long n)
{
  long i = (long)blockIdx.x*blockDim.x + threadIdx.x;
  if (i < n) { float r = fmaxf(g[i], 0.f); g[i] = r*r*u[i]; }
}

extern "C" void kernel_launch(void* const* d_in, const int* in_sizes, int n_in,
                              void* d_out, int out_size, void* d_ws, size_t ws_size,
                              hipStream_t stream)
{
  const float* hidden   = (const float*)d_in[0];
  const float* cosb     = (const float*)d_in[1];
  const float* sinb     = (const float*)d_in[2];
  const float* w_in_ln  = (const float*)d_in[3];
  const float* w_q      = (const float*)d_in[4];
  const float* w_k      = (const float*)d_in[5];
  const float* w_v      = (const float*)d_in[6];
  const float* w_o      = (const float*)d_in[7];
  const float* w_asub   = (const float*)d_in[8];
  const float* w_pln    = (const float*)d_in[9];
  const float* w_gate   = (const float*)d_in[10];
  const float* w_up     = (const float*)d_in[11];
  const float* w_fsub   = (const float*)d_in[12];
  const float* w_down   = (const float*)d_in[13];
  float* out = (float*)d_out;

  char* cur = (char*)d_ws;
  auto alloc = [&](size_t bytes) -> void* {
    void* r = cur; cur += (bytes + 255) & ~(size_t)255; return r;
  };

  float*  acc   = (float*)alloc(256);
  int8_t* wqb[7];
  const float* Wf[7] = { w_q, w_k, w_v, w_o, w_gate, w_up, w_down };
  const long   Wn[7] = { (long)NH_*HD_*H_, (long)NKV_*HD_*H_, (long)NKV_*HD_*H_,
                         (long)H_*NH_*HD_, (long)FF_*H_, (long)FF_*H_, (long)H_*FF_ };
  for (int i = 0; i < 7; i++) wqb[i] = (int8_t*)alloc((size_t)Wn[i]);

  int8_t* xq    = (int8_t*)alloc((size_t)S_*FF_);
  float*  srow  = (float*) alloc((size_t)S_*4);
  float*  qf    = (float*) alloc((size_t)S_*NH_*HD_*4);
  float*  kf    = (float*) alloc((size_t)S_*NKV_*HD_*4);
  float*  vf    = (float*) alloc((size_t)S_*NKV_*HD_*4);
  _Float16* q16 = (_Float16*)alloc((size_t)NH_*S_*HD_*2);
  _Float16* k16 = (_Float16*)alloc((size_t)NKV_*S_*HD_*2);
  _Float16* v16 = (_Float16*)alloc((size_t)NKV_*HD_*S_*2);
  float*  oproj = (float*) alloc((size_t)S_*H_*4);
  float*  hbuf  = (float*) alloc((size_t)S_*H_*4);
  float*  gbuf  = (float*) alloc((size_t)S_*FF_*4);
  float*  ubuf  = (float*) alloc((size_t)S_*FF_*4);
  (void)ws_size; (void)n_in; (void)in_sizes; (void)out_size;

  (void)hipMemsetAsync(acc, 0, 256, stream);
  for (int i = 0; i < 7; i++) {
    long n = Wn[i];
    int g = (int)(((n + 255) / 256) < 2048 ? ((n + 255) / 256) : 2048);
    k_abssum<<<g, 256, 0, stream>>>(Wf[i], n, acc + i);
    k_wquant<<<g, 256, 0, stream>>>(Wf[i], n, acc + i, 1.f/(float)n, wqb[i]);
  }

  auto gemm = [&](const int8_t* A, const int8_t* W, float* ws_acc, float invn,
                  float* o, int M, int N, int K) {
    int blocks = (M/128)*(N/64);
    k_gemm_q8<<<blocks, 256, 0, stream>>>(A, srow, W, ws_acc, invn, o, M, N, K);
  };

  k_rmsq<<<S_, 256, 0, stream>>>(hidden, w_in_ln, H_, xq, srow);
  gemm(xq, wqb[0], acc+0, 1.f/(float)Wn[0], qf, S_, NH_*HD_,  H_);
  gemm(xq, wqb[1], acc+1, 1.f/(float)Wn[1], kf, S_, NKV_*HD_, H_);
  gemm(xq, wqb[2], acc+2, 1.f/(float)Wn[2], vf, S_, NKV_*HD_, H_);

  {
    long nq = (long)NH_*S_*HD_,  nk = (long)NKV_*S_*HD_;
    k_rope<<<(int)((nq+255)/256), 256, 0, stream>>>(qf, cosb, sinb, q16, NH_);
    k_rope<<<(int)((nk+255)/256), 256, 0, stream>>>(kf, cosb, sinb, k16, NKV_);
    k_vt  <<<(int)((nk+255)/256), 256, 0, stream>>>(vf, v16);
  }

  float* attnO = qf;
  k_flash<<<(S_/16)*NH_/8, 256, 0, stream>>>(q16, k16, v16, attnO);

  k_rmsq<<<S_, 256, 0, stream>>>(attnO, w_asub, H_, xq, srow);
  gemm(xq, wqb[3], acc+3, 1.f/(float)Wn[3], oproj, S_, H_, NH_*HD_);
  {
    long n = (long)S_*H_;
    k_add<<<(int)((n+255)/256), 256, 0, stream>>>(hidden, oproj, hbuf, n);
  }

  k_rmsq<<<S_, 256, 0, stream>>>(hbuf, w_pln, H_, xq, srow);
  gemm(xq, wqb[4], acc+4, 1.f/(float)Wn[4], gbuf, S_, FF_, H_);
  gemm(xq, wqb[5], acc+5, 1.f/(float)Wn[5], ubuf, S_, FF_, H_);
  {
    long n = (long)S_*FF_;
    k_sqrelu_mul<<<(int)((n+255)/256), 256, 0, stream>>>(gbuf, ubuf, n);
  }
  k_rmsq<<<S_, 256, 0, stream>>>(gbuf, w_fsub, FF_, xq, srow);
  gemm(xq, wqb[6], acc+6, 1.f/(float)Wn[6], oproj, S_, H_, FF_);
  {
    long n = (long)S_*H_;
    k_add<<<(int)((n+255)/256), 256, 0, stream>>>(hbuf, oproj, out, n);
  }
}
